// GRANDE_82566451299124
// MI455X (gfx1250) — matrix operand order repaired and checked by reading, not compiled, not run on hardware
//
#include <hip/hip_runtime.h>
#include <hip/hip_bf16.h>

typedef __attribute__((ext_vector_type(16))) _Float16 v16h;
typedef __attribute__((ext_vector_type(8)))  float    v8f;

#define WMMA_F16(a, b, c) \
  __builtin_amdgcn_wmma_f32_16x16x32_f16(false, (a), false, (b), (short)0, (c), false, false)

constexpr int kB = 8192;
constexpr int kT = 128;
constexpr int kD = 7;
constexpr int kI = 128;
constexpr int kC = 10;
constexpr int kH = 64;
constexpr int kL = 128;   // 2^kD

// B-fragment layout (16x16x32 f16 WMMA, wave32):
//   fragment f = [chunk c][lane L][half e],  k = c*32 + (L>>4)*8 + (e&7) + (e>>3)*16,  n = L&15
//   stored as 16 contiguous halves (32 B) per lane -> one aligned v16h load.

// ---------------------------------------------------------------------------
// Pre-pass A: fmask -> f16 B-fragments.  dst[t][c][lane][e]
// ---------------------------------------------------------------------------
__global__ __launch_bounds__(256) void swz_fmask(const float* __restrict__ fm,
                                                 _Float16* __restrict__ dst) {
  const int idx = blockIdx.x * 256 + threadIdx.x;
  if (idx >= kT * 4 * 32 * 16) return;
  const int e = idx & 15, lane = (idx >> 4) & 31, c = (idx >> 9) & 3, t = idx >> 11;
  const int n = lane & 15;
  const int k = c * 32 + (lane >> 4) * 8 + (e & 7) + (e >> 3) * 16;
  dst[idx] = (n < kD) ? (_Float16)fm[((size_t)t * kD + n) * kI + k] : (_Float16)0.f;
}

// ---------------------------------------------------------------------------
// Pre-pass B: softmax(leaf_outputs) -> f16 B-fragments. dst[t][c][lane][e]
// One thread per (t, leaf-row k); scatters its 16 column slots.
// ---------------------------------------------------------------------------
__global__ __launch_bounds__(256) void leaf_softmax_swz(
    const float* __restrict__ leaf, _Float16* __restrict__ lpB) {
  const int row = blockIdx.x * 256 + threadIdx.x;  // t*kL + k
  if (row >= kT * kL) return;
  const int t = row >> 7, k = row & (kL - 1);
  const float* src = leaf + (size_t)row * kC;
  float mx = src[0];
#pragma unroll
  for (int c = 1; c < kC; ++c) mx = fmaxf(mx, src[c]);
  float p[kC], sum = 0.f;
#pragma unroll
  for (int c = 0; c < kC; ++c) { p[c] = __expf(src[c] - mx); sum += p[c]; }
  const float inv = 1.f / sum;
  const int c  = k >> 5, hi = (k >> 3) & 1, e = (k & 7) | ((k >> 1) & 8);
  _Float16* base = lpB + ((((size_t)t * 4 + c) * 32 + hi * 16) * 16 + e);
#pragma unroll
  for (int n = 0; n < 16; ++n)
    base[(size_t)n * 16] = (n < kC) ? (_Float16)(p[n] * inv) : (_Float16)0.f;
}

// ---------------------------------------------------------------------------
// Pre-pass C: W1 (I x H) -> f16 B-fragments.  dst[nt][c][lane][e], nt<4, c<4
// ---------------------------------------------------------------------------
__global__ __launch_bounds__(256) void swz_w1(const float* __restrict__ W1,
                                              _Float16* __restrict__ dst) {
  const int idx = blockIdx.x * 256 + threadIdx.x;
  if (idx >= 4 * 4 * 32 * 16) return;
  const int e = idx & 15, lane = (idx >> 4) & 31, c = (idx >> 9) & 3, nt = idx >> 11;
  const int k = c * 32 + (lane >> 4) * 8 + (e & 7) + (e >> 3) * 16;
  const int n = nt * 16 + (lane & 15);
  dst[idx] = (_Float16)W1[(size_t)k * kH + n];
}

// ---------------------------------------------------------------------------
// Pre-pass D: W2 (H x T) -> f16 B-fragments.  dst[nt][c][lane][e], nt<8, c<2
// ---------------------------------------------------------------------------
__global__ __launch_bounds__(256) void swz_w2(const float* __restrict__ W2,
                                              _Float16* __restrict__ dst) {
  const int idx = blockIdx.x * 256 + threadIdx.x;
  if (idx >= 8 * 2 * 32 * 16) return;
  const int e = idx & 15, lane = (idx >> 4) & 31, c = (idx >> 9) & 1, nt = idx >> 10;
  const int k = c * 32 + (lane >> 4) * 8 + (e & 7) + (e >> 3) * 16;
  const int n = nt * 16 + (lane & 15);
  dst[idx] = (_Float16)W2[(size_t)k * kT + n];
}

// ---------------------------------------------------------------------------
// Kernel: attn = softmax(relu(x@W1+b1)@W2+b2) over T.  One wave / 16 rows.
// ---------------------------------------------------------------------------
__global__ __launch_bounds__(32) void grande_attn(
    const float* __restrict__ x, const v16h* __restrict__ W1B,
    const float* __restrict__ b1, const v16h* __restrict__ W2B,
    const float* __restrict__ b2, float* __restrict__ attn) {
  __shared__ float h_lds[16][kH];     // 4 KB
  __shared__ float l_lds[16][kT];     // 8 KB
  const int lane = threadIdx.x;
  const int m  = lane & 15;
  const int hi = lane >> 4;
  const int b0 = blockIdx.x * 16;

  // A-fragments of x tile (16 x 128 f32 -> f16), 4 K-chunks of 32.
  v16h xa[4];
  const float* xrow = x + (size_t)(b0 + m) * kI;
#pragma unroll
  for (int c = 0; c < 4; ++c) {
    const int kb = c * 32 + hi * 8;
#pragma unroll
    for (int j = 0; j < 8; ++j) {
      xa[c][j]     = (_Float16)xrow[kb + j];
      xa[c][8 + j] = (_Float16)xrow[kb + 16 + j];
    }
  }

  // Layer 1: h = relu(x@W1 + b1), H=64 -> 4 N-tiles.
#pragma unroll
  for (int nt = 0; nt < 4; ++nt) {
    v8f hacc = {};
#pragma unroll
    for (int c = 0; c < 4; ++c)
      hacc = WMMA_F16(xa[c], W1B[((size_t)nt * 4 + c) * 32 + lane], hacc);
    const int n = nt * 16 + m;
    const float bias = b1[n];
#pragma unroll
    for (int r = 0; r < 8; ++r) {
      const float hv = hacc[r] + bias;
      h_lds[r + hi * 8][n] = hv > 0.f ? hv : 0.f;
    }
  }
  __syncthreads();

  // A-fragments of h (16 x 64), 2 K-chunks.
  v16h ha[2];
#pragma unroll
  for (int c = 0; c < 2; ++c) {
    const int kb = c * 32 + hi * 8;
#pragma unroll
    for (int j = 0; j < 8; ++j) {
      ha[c][j]     = (_Float16)h_lds[m][kb + j];
      ha[c][8 + j] = (_Float16)h_lds[m][kb + 16 + j];
    }
  }

  // Layer 2: logits = h@W2 + b2, T=128 -> 8 N-tiles.
#pragma unroll
  for (int nt = 0; nt < 8; ++nt) {
    v8f lacc = {};
#pragma unroll
    for (int c = 0; c < 2; ++c)
      lacc = WMMA_F16(ha[c], W2B[((size_t)nt * 2 + c) * 32 + lane], lacc);
    const int n = nt * 16 + m;
    const float bias = b2[n];
#pragma unroll
    for (int r = 0; r < 8; ++r) l_lds[r + hi * 8][n] = lacc[r] + bias;
  }
  __syncthreads();

  // Row softmax over T (lanes 0..15, one row each).
  if (lane < 16) {
    float mx = -1e30f;
    for (int t = 0; t < kT; ++t) mx = fmaxf(mx, l_lds[lane][t]);
    float sum = 0.f;
    for (int t = 0; t < kT; ++t) {
      const float e = __expf(l_lds[lane][t] - mx);
      l_lds[lane][t] = e;
      sum += e;
    }
    const float inv = 1.f / sum;
    float* dst = attn + (size_t)(b0 + lane) * kT;
    for (int t = 0; t < kT; ++t) dst[t] = l_lds[lane][t] * inv;
  }
}

// ---------------------------------------------------------------------------
// Kernel: fused trees.  One wave owns 16 rows of b, loops all 128 trees,
// accumulating attn-weighted (p @ leaf_probs) in one WMMA accumulator.
// ---------------------------------------------------------------------------
__global__ __launch_bounds__(32) void grande_trees(
    const float* __restrict__ x, const v16h* __restrict__ fmB,
    const float* __restrict__ thr, const v16h* __restrict__ lpB,
    const float* __restrict__ attn, float* __restrict__ out) {
  __shared__ float s_lds[16][8];
  const int lane = threadIdx.x;
  const int m  = lane & 15;
  const int hi = lane >> 4;
  const int b0 = blockIdx.x * 16;

  // x A-fragments (reused across all trees): 4 K-chunks of 32.
  v16h xa[4];
  const float* xrow = x + (size_t)(b0 + m) * kI;
#pragma unroll
  for (int c = 0; c < 4; ++c) {
    const int kb = c * 32 + hi * 8;
#pragma unroll
    for (int j = 0; j < 8; ++j) {
      xa[c][j]     = (_Float16)xrow[kb + j];
      xa[c][8 + j] = (_Float16)xrow[kb + 16 + j];
    }
  }

  v8f acc = {};  // 16x16, columns 0..9 valid; accumulates over all trees

  for (int t = 0; t < kT; ++t) {
    if (t + 1 < kT) {  // pull next tree's fragments toward L2/L0
      __builtin_prefetch((const void*)(fmB + ((size_t)(t + 1) * 4) * 32 + lane), 0, 1);
      __builtin_prefetch((const void*)(lpB + ((size_t)(t + 1) * 4) * 32 + lane), 0, 1);
    }

    // ---- z(16x7) = x_tile @ fmask[t]^T : 4 WMMAs, pre-swizzled B frags
    v8f zc = {};
#pragma unroll
    for (int c = 0; c < 4; ++c)
      zc = WMMA_F16(xa[c], fmB[((size_t)t * 4 + c) * 32 + lane], zc);

    // ---- s = 0.5*(z'/(1+|z'|)+1), z' = z - thr ; stash in LDS
    if (m < kD) {
      const float th = thr[t * kD + m];
#pragma unroll
      for (int r = 0; r < 8; ++r) {
        const float zz = zc[r] - th;
        s_lds[r + hi * 8][m] = 0.5f * (zz * __builtin_amdgcn_rcpf(1.0f + fabsf(zz)) + 1.0f);
      }
    }
    __syncthreads();

    float srow[kD];
#pragma unroll
    for (int d = 0; d < kD; ++d) srow[d] = s_lds[m][d];
    const float ascale = attn[(size_t)(b0 + m) * kT + t];

    // ---- p A-fragments (attn-scaled), tree-structured prefix products.
#pragma unroll
    for (int c = 0; c < 4; ++c) {
      v16h pa;
#pragma unroll
      for (int g = 0; g < 2; ++g) {
        const int lbase = c * 32 + hi * 8 + g * 16;   // 8 consecutive leaves
        float pref = ascale;
#pragma unroll
        for (int d = 3; d < kD; ++d)
          pref *= ((lbase >> d) & 1) ? (1.0f - srow[d]) : srow[d];
#pragma unroll
        for (int j = 0; j < 8; ++j) {
          const int l = lbase + j;
          float prod = pref;
#pragma unroll
          for (int d = 0; d < 3; ++d)
            prod *= ((l >> d) & 1) ? (1.0f - srow[d]) : srow[d];
          pa[g * 8 + j] = (_Float16)prod;
        }
      }
      acc = WMMA_F16(pa, lpB[((size_t)t * 4 + c) * 32 + lane], acc);
    }
    __syncthreads();  // s_lds reused next tree
  }

  // ---- store out (16 x 10)
  if (m < kC) {
#pragma unroll
    for (int r = 0; r < 8; ++r)
      out[(size_t)(b0 + r + hi * 8) * kC + m] = acc[r];
  }
}

// ---------------------------------------------------------------------------
extern "C" void kernel_launch(void* const* d_in, const int* in_sizes, int n_in,
                              void* d_out, int out_size, void* d_ws, size_t ws_size,
                              hipStream_t stream) {
  const float* x     = (const float*)d_in[0];  // B x I
  const float* fmask = (const float*)d_in[1];  // T x D x I
  const float* thr   = (const float*)d_in[2];  // T x D
  const float* leaf  = (const float*)d_in[3];  // T x L x C
  const float* W1    = (const float*)d_in[4];  // I x H
  const float* b1    = (const float*)d_in[5];  // H
  const float* W2    = (const float*)d_in[6];  // H x T
  const float* b2    = (const float*)d_in[7];  // T
  float* out = (float*)d_out;                  // B x C

  // Workspace layout (all regions 32B-aligned):
  //   [attn f32 B*T = 4 MB][fmB f16 512 KB][lpB f16 512 KB][W1B 16 KB][W2B 16 KB]
  char* ws = (char*)d_ws;
  float*    attn_ws = (float*)ws;
  _Float16* fmB     = (_Float16*)(ws + (size_t)kB * kT * 4);
  _Float16* lpB     = fmB + (size_t)kT * 4 * 32 * 16;
  _Float16* W1B     = lpB + (size_t)kT * 4 * 32 * 16;
  _Float16* W2B     = W1B + (size_t)4 * 4 * 32 * 16;

  swz_fmask<<<(kT * 4 * 32 * 16 + 255) / 256, 256, 0, stream>>>(fmask, fmB);
  leaf_softmax_swz<<<(kT * kL + 255) / 256, 256, 0, stream>>>(leaf, lpB);
  swz_w1<<<(4 * 4 * 32 * 16 + 255) / 256, 256, 0, stream>>>(W1, W1B);
  swz_w2<<<(8 * 2 * 32 * 16 + 255) / 256, 256, 0, stream>>>(W2, W2B);
  grande_attn<<<kB / 16, 32, 0, stream>>>(x, (const v16h*)W1B, b1, (const v16h*)W2B,
                                          b2, attn_ws);
  grande_trees<<<kB / 16, 32, 0, stream>>>(x, (const v16h*)fmB, thr, (const v16h*)lpB,
                                           attn_ws, out);
}
